// LSTMEncoder_31645319037020
// MI455X (gfx1250) — matrix operand order repaired and checked by reading, not compiled, not run on hardware
//
#include <hip/hip_runtime.h>

// ---------------------------------------------------------------------------
// LSTM encoder for MI455X (gfx1250, wave32, WMMA).
//   VOCAB=30522, E=128, H=256, B=128, T=1024, out=[128,512] f32
// ---------------------------------------------------------------------------

typedef __bf16 bf16;
typedef __attribute__((ext_vector_type(16))) __bf16 v16bf;
typedef __attribute__((ext_vector_type(8)))  float  v8f;
typedef __attribute__((ext_vector_type(2)))  float  v2f;
typedef __attribute__((ext_vector_type(4)))  int    v4i;

#define T_STEPS 1024
#define BATCH   128
#define EMBED   128
#define HID     256
#define KDIM    384            // [h(256) | x(128)] fused K
#define KP      392            // padded LDS row stride (bf16 elems): 784B, breaks bank conflicts

// ---- workspace layout (bytes) ----
static const size_t OFF_SYNC = 0;                                  // 256 B sync counters
static const size_t OFF_HBUF = 256;                                // [2 dir][2 buf][128][256] bf16
static const size_t SZ_HBUF  = (size_t)2 * 2 * BATCH * HID * 2;    // 262144
static const size_t OFF_HFIN = OFF_HBUF + SZ_HBUF;                 // [2][128][256] f32 final h
static const size_t SZ_HFIN  = (size_t)2 * BATCH * HID * 4;        // 262144
static const size_t OFF_WC   = OFF_HFIN + SZ_HFIN;                 // [2][1024][384] bf16 fused weights
static const size_t SZ_WC    = (size_t)2 * 1024 * KDIM * 2;        // 1572864
static const size_t OFF_EMB  = OFF_WC + SZ_WC;                     // [T][B][E] bf16 time-major
static const size_t LDS_BYTES = (size_t)(256 + 128) * KP * 2;      // 301056 < 320KB/WGP

// ---------------------------------------------------------------------------
// Fast branchless activations: CDNA5 v_tanh_f32 when available.
// ---------------------------------------------------------------------------
#if __has_builtin(__builtin_amdgcn_tanhf)
static __device__ __forceinline__ float fast_tanh(float x) {
    return __builtin_amdgcn_tanhf(x);
}
static __device__ __forceinline__ float fast_sig(float x) {
    return fmaf(0.5f, __builtin_amdgcn_tanhf(0.5f * x), 0.5f);
}
#else
static __device__ __forceinline__ float fast_tanh(float x) {
    // branchless: tanh(x) = 2/(1+exp(-2x)) - 1, saturates correctly at +-inf
    return fmaf(2.f, __builtin_amdgcn_rcpf(1.f + __expf(-2.f * x)), -1.f);
}
static __device__ __forceinline__ float fast_sig(float x) {
    return __builtin_amdgcn_rcpf(1.f + __expf(-x));
}
#endif

// ---------------------------------------------------------------------------
// Async global->LDS copy (CDNA5 ASYNCcnt path), guarded with fallback.
//   Probe showed: arg0 = int4 * in global (__device__) address space.
// ---------------------------------------------------------------------------
#if __has_builtin(__builtin_amdgcn_global_load_async_to_lds_b128) && \
    __has_builtin(__builtin_amdgcn_s_wait_asynccnt)
#define USE_ASYNC_LDS 1
typedef __attribute__((address_space(1))) v4i v4i_g;   // global
typedef __attribute__((address_space(3))) v4i v4i_l;   // LDS
static __device__ __forceinline__ void cp16_async(void* lds_dst, const void* gsrc) {
    __builtin_amdgcn_global_load_async_to_lds_b128(
        (v4i_g*)(v4i*)gsrc, (v4i_l*)(v4i*)lds_dst, 0, 0);
}
static __device__ __forceinline__ void cp_wait() {
    __builtin_amdgcn_s_wait_asynccnt(0);
}
#else
#define USE_ASYNC_LDS 0
static __device__ __forceinline__ void cp16_async(void* lds_dst, const void* gsrc) {
    *(uint4*)lds_dst = *(const uint4*)gsrc;
}
static __device__ __forceinline__ void cp_wait() {}
#endif

// ---------------------------------------------------------------------------
// Kernel 1: embedding gather -> time-major bf16 (padding_idx=0 -> zeros)
// ---------------------------------------------------------------------------
__global__ void __launch_bounds__(256)
embed_kernel(const int* __restrict__ ids, const float* __restrict__ table,
             bf16* __restrict__ emb)
{
    int gid   = blockIdx.x * 256 + threadIdx.x;  // T*B*32 threads
    int chunk = gid & 31;                        // 4 floats per thread
    int row   = gid >> 5;                        // row = t*128 + b
    int t = row >> 7, b = row & 127;
    int id = ids[b * T_STEPS + t];
    float4 v = make_float4(0.f, 0.f, 0.f, 0.f);
    if (id != 0)
        v = *(const float4*)(table + (size_t)id * EMBED + chunk * 4);
    union { bf16 h[4]; uint2 u; } pk;
    pk.h[0] = (bf16)v.x; pk.h[1] = (bf16)v.y; pk.h[2] = (bf16)v.z; pk.h[3] = (bf16)v.w;
    *(uint2*)(emb + (size_t)row * EMBED + chunk * 4) = pk.u;
}

// ---------------------------------------------------------------------------
// Kernel 2: pack fused recurrent weights Wc[d][r][k] = k<256 ? w_hh[r][k] : w_ih[r][k-256]
// ---------------------------------------------------------------------------
__global__ void __launch_bounds__(256)
wpack_kernel(const float* __restrict__ wih_f, const float* __restrict__ whh_f,
             const float* __restrict__ wih_r, const float* __restrict__ whh_r,
             bf16* __restrict__ Wc)
{
    int gid = blockIdx.x * 256 + threadIdx.x;    // 2*1024*384
    if (gid >= 2 * 1024 * KDIM) return;
    int d = gid / (1024 * KDIM);
    int rem = gid % (1024 * KDIM);
    int r = rem / KDIM, k = rem % KDIM;
    const float* wih = d ? wih_r : wih_f;
    const float* whh = d ? whh_r : whh_f;
    float v = (k < HID) ? whh[r * HID + k] : wih[r * EMBED + (k - HID)];
    Wc[gid] = (bf16)v;
}

// ---------------------------------------------------------------------------
// Kernel 3: persistent bidirectional LSTM scan.
//   grid = 8 blocks: d = blk>>2 (direction), wg = blk&3 (64 h-cols slice)
//   block = 256 threads = 8 waves; wave (wm in 0..1, wcb in 0..3):
//     wave tile = M rows [64*wm, 64*wm+64) x cols [wg*64+wcb*16, +16) x 4 gates
// ---------------------------------------------------------------------------
__global__ void __launch_bounds__(256)
lstm_scan_kernel(const bf16* __restrict__ emb,   // [T][B][E]
                 const bf16* __restrict__ Wc,    // [2][1024][KDIM]
                 const float* __restrict__ bF,
                 const float* __restrict__ bR,
                 bf16* __restrict__ hbuf,        // [2][2][B][H]
                 float* __restrict__ hfin,       // [2][B][H]
                 unsigned int* __restrict__ sync)
{
    extern __shared__ char smem_raw[];
    bf16* Wl  = (bf16*)smem_raw;                 // [256][KP]  weight slice
    bf16* Xin = Wl + 256 * KP;                   // [128][KP]  [h | x_t]

    const int tid  = threadIdx.x;
    const int d    = blockIdx.x >> 2;
    const int wg   = blockIdx.x & 3;
    const int wave = tid >> 5;
    const int lane = tid & 31;
    const int wm   = wave >> 2;                  // 0..1
    const int wcb  = wave & 3;                   // 0..3
    const int nloc = lane & 15;
    const int half = lane >> 4;
    const int ncol = wg * 64 + wcb * 16 + nloc;  // this lane's h column

    // ---- stage W slice once: local row q*64+rr <- global row q*256 + wg*64 + rr
    const bf16* W = Wc + (size_t)d * 1024 * KDIM;
    for (int i = tid; i < 256 * (KDIM / 8); i += 256) {
        int row = i / (KDIM / 8), col8 = i % (KDIM / 8);
        int q = row >> 6, rr = row & 63;
        cp16_async(Wl + row * KP + col8 * 8,
                   W + ((q * 256 + wg * 64 + rr) * KDIM + col8 * 8));
    }

    const float* bias = (d == 0) ? bF : bR;
    float bq[4];
#pragma unroll
    for (int q = 0; q < 4; ++q) bq[q] = bias[q * HID + ncol];

    v8f c[4];                                    // cell state, persistent in regs
#pragma unroll
    for (int mt = 0; mt < 4; ++mt)
#pragma unroll
        for (int e = 0; e < 8; ++e) c[mt][e] = 0.f;

    bf16* hb = hbuf + (size_t)d * 2 * BATCH * HID;
    cp_wait();
    __syncthreads();

    for (int t = 0; t < T_STEPS; ++t) {
        // ---- stage [h(t-1) | x_t] into LDS (async DMA when available)
        const bf16* hsrc = hb + (size_t)(t & 1) * BATCH * HID;
        for (int i = tid; i < BATCH * (HID / 8); i += 256) {
            int row = i >> 5, col8 = i & 31;
            cp16_async(Xin + row * KP + col8 * 8, hsrc + row * HID + col8 * 8);
        }
        int tsrc = d ? (T_STEPS - 1 - t) : t;
        const bf16* xsrc = emb + (size_t)tsrc * BATCH * EMBED;
        for (int i = tid; i < BATCH * (EMBED / 8); i += 256) {
            int row = i >> 4, col8 = i & 15;
            cp16_async(Xin + row * KP + HID + col8 * 8, xsrc + row * EMBED + col8 * 8);
        }
        cp_wait();
        __syncthreads();

        // ---- gates = [h|x] * W^T + b via bf16 WMMA, f32 accumulate
        v8f acc[4][4];
#pragma unroll
        for (int q = 0; q < 4; ++q) {
            v8f a;
#pragma unroll
            for (int e = 0; e < 8; ++e) a[e] = bq[q];
#pragma unroll
            for (int mt = 0; mt < 4; ++mt) acc[q][mt] = a;
        }

        for (int k = 0; k < KDIM; k += 32) {
            v16bf bfrag[4];
#pragma unroll
            for (int q = 0; q < 4; ++q) {
                // B 32x16 bf16: lane = col n, K split (0..7,16..23 | 8..15,24..31)
                const bf16* bp = Wl + (q * 64 + wcb * 16 + nloc) * KP + k + half * 8;
                union { uint4 u[2]; v16bf v; } fb;
                fb.u[0] = *(const uint4*)(bp);
                fb.u[1] = *(const uint4*)(bp + 16);
                bfrag[q] = fb.v;
            }
#pragma unroll
            for (int mt = 0; mt < 4; ++mt) {
                int m0 = (wm * 4 + mt) * 16;
                // A 16x32 bf16: lane = row m, K split (0..7,16..23 | 8..15,24..31)
                const bf16* ap = Xin + (m0 + nloc) * KP + k + half * 8;
                union { uint4 u[2]; v16bf v; } fa;
                fa.u[0] = *(const uint4*)(ap);
                fa.u[1] = *(const uint4*)(ap + 16);
                v16bf afrag = fa.v;
#pragma unroll
                for (int q = 0; q < 4; ++q)
                    acc[q][mt] = __builtin_amdgcn_wmma_f32_16x16x32_bf16(
                        false, afrag, false, bfrag[q], (short)0, acc[q][mt],
                        false, false);
            }
        }

        // ---- cell update entirely in C-layout registers (lane: col ncol, rows r+8*half)
        bf16* hdst = hb + (size_t)((t + 1) & 1) * BATCH * HID;
#pragma unroll
        for (int mt = 0; mt < 4; ++mt) {
            int m0 = (wm * 4 + mt) * 16;
#pragma unroll
            for (int r = 0; r < 8; ++r) {
                float iv = fast_sig(acc[0][mt][r]);
                float fv = fast_sig(acc[1][mt][r]);
                float gv = fast_tanh(acc[2][mt][r]);
                float ov = fast_sig(acc[3][mt][r]);
                float cv = fmaf(fv, c[mt][r], iv * gv);
                c[mt][r] = cv;
                float hv = ov * fast_tanh(cv);
                int row = m0 + r + half * 8;
                hdst[row * HID + ncol] = (bf16)hv;
                if (t == T_STEPS - 1)
                    hfin[(size_t)d * BATCH * HID + row * HID + ncol] = hv;
            }
        }

        // ---- cross-WG step barrier (4 WGs per direction, monotonic counter)
        __threadfence();
        __syncthreads();
        if (tid == 0) {
            __hip_atomic_fetch_add(&sync[d], 1u, __ATOMIC_RELEASE,
                                   __HIP_MEMORY_SCOPE_AGENT);
            unsigned target = 4u * (unsigned)(t + 1);
            while (__hip_atomic_load(&sync[d], __ATOMIC_ACQUIRE,
                                     __HIP_MEMORY_SCOPE_AGENT) < target)
                __builtin_amdgcn_s_sleep(2);
        }
        __syncthreads();
    }
}

// ---------------------------------------------------------------------------
// Kernel 4: out = relu([h_f | h_r] * proj_w^T + proj_b) with f32 WMMA 16x16x4.
//   grid 16 x 64 threads (2 waves); wave owns one N-tile, all 8 M-tiles.
// ---------------------------------------------------------------------------
__global__ void __launch_bounds__(64)
proj_kernel(const float* __restrict__ hfin,  // [2][128][256]
            const float* __restrict__ pw,    // [512][512]
            const float* __restrict__ pb,    // [512]
            float* __restrict__ out)         // [128][512]
{
    int tid  = threadIdx.x;
    int lane = tid & 31, wave = tid >> 5;
    int n0   = (blockIdx.x * 2 + wave) * 16;
    int nloc = lane & 15, half = lane >> 4;

    v8f acc[8];
#pragma unroll
    for (int mt = 0; mt < 8; ++mt)
#pragma unroll
        for (int e = 0; e < 8; ++e) acc[mt][e] = 0.f;

    for (int k = 0; k < 512; k += 4) {
        int kk = k + half * 2;
        const float* bp = pw + (size_t)(n0 + nloc) * 512 + kk;
        v2f bfrag; bfrag[0] = bp[0]; bfrag[1] = bp[1];
#pragma unroll
        for (int mt = 0; mt < 8; ++mt) {
            int row = mt * 16 + nloc;
            const float* ap = (kk < HID)
                ? hfin + (size_t)row * HID + kk
                : hfin + (size_t)BATCH * HID + (size_t)row * HID + (kk - HID);
            v2f afrag; afrag[0] = ap[0]; afrag[1] = ap[1];
            acc[mt] = __builtin_amdgcn_wmma_f32_16x16x4_f32(
                false, afrag, false, bfrag, (short)0, acc[mt], false, false);
        }
    }

    float bb = pb[n0 + nloc];
#pragma unroll
    for (int mt = 0; mt < 8; ++mt)
#pragma unroll
        for (int r = 0; r < 8; ++r) {
            int row = mt * 16 + r + half * 8;
            out[(size_t)row * 512 + (n0 + nloc)] = fmaxf(acc[mt][r] + bb, 0.f);
        }
}

// ---------------------------------------------------------------------------
extern "C" void kernel_launch(void* const* d_in, const int* in_sizes, int n_in,
                              void* d_out, int out_size, void* d_ws, size_t ws_size,
                              hipStream_t stream)
{
    const int*   ids   = (const int*)  d_in[0];
    const float* table = (const float*)d_in[2];
    const float* wih_f = (const float*)d_in[3];
    const float* whh_f = (const float*)d_in[4];
    const float* b_f   = (const float*)d_in[5];
    const float* wih_r = (const float*)d_in[6];
    const float* whh_r = (const float*)d_in[7];
    const float* b_r   = (const float*)d_in[8];
    const float* pw    = (const float*)d_in[9];
    const float* pb    = (const float*)d_in[10];
    float* out = (float*)d_out;

    char* ws = (char*)d_ws;
    unsigned* sync = (unsigned*)(ws + OFF_SYNC);
    bf16*  hbuf = (bf16*) (ws + OFF_HBUF);
    float* hfin = (float*)(ws + OFF_HFIN);
    bf16*  Wc   = (bf16*) (ws + OFF_WC);
    bf16*  emb  = (bf16*) (ws + OFF_EMB);

    // zero sync counters + h double-buffers (deterministic per launch)
    (void)hipMemsetAsync(ws, 0, OFF_HFIN, stream);

    embed_kernel<<<(T_STEPS * BATCH * 32) / 256, 256, 0, stream>>>(ids, table, emb);
    wpack_kernel<<<(2 * 1024 * KDIM + 255) / 256, 256, 0, stream>>>(
        wih_f, whh_f, wih_r, whh_r, Wc);

    (void)hipFuncSetAttribute((const void*)lstm_scan_kernel,
                              hipFuncAttributeMaxDynamicSharedMemorySize,
                              (int)LDS_BYTES);
    lstm_scan_kernel<<<8, 256, LDS_BYTES, stream>>>(emb, Wc, b_f, b_r,
                                                    hbuf, hfin, sync);

    proj_kernel<<<16, 64, 0, stream>>>(hfin, pw, pb, out);
}
